// LSTMFromScratch_6751688589450
// MI455X (gfx1250) — matrix operand order repaired and checked by reading, not compiled, not run on hardware
//
#include <hip/hip_runtime.h>
#include <hip/hip_fp16.h>

// Problem dims (match reference)
#define SEQ     512
#define BATCH   128
#define IN_DIM  256
#define HID     512
#define OUT_DIM 256

#define NBLK    32          // persistent workgroups
#define NTHR    256         // 8 waves of 32
#define NTHREADS_TOTAL (NBLK * NTHR)

// LDS B-panel: 64 gate columns x 32 k, row stride padded to 40 halves (80B)
// -> lanes 0..15 map to 16 distinct bank groups (20*n mod 64 all distinct).
#define BROW_STRIDE 40
#define BPANEL      (64 * BROW_STRIDE)

typedef __attribute__((ext_vector_type(16))) _Float16 v16h;
typedef __attribute__((ext_vector_type(8)))  float    v8f;
typedef __attribute__((ext_vector_type(4)))  int      v4i;

typedef __attribute__((address_space(1))) v4i* gptr_v4i;
typedef __attribute__((address_space(3))) v4i* lptr_v4i;

union TileU {
  v16h v;
  uint4 q[2];
};

#if __has_builtin(__builtin_amdgcn_global_load_async_to_lds_b128)
#define HAVE_ASYNC_LDS 1
#endif

// Copy 16B global -> LDS. Async path uses the CDNA5 async DMA (ASYNCcnt),
// fallback is a plain synchronous LDS store.
__device__ __forceinline__ void copy_b128_to_lds(_Float16* lds_dst,
                                                 const _Float16* gsrc) {
#ifdef HAVE_ASYNC_LDS
  __builtin_amdgcn_global_load_async_to_lds_b128(
      (gptr_v4i)(void*)gsrc, (lptr_v4i)(void*)lds_dst, 0, 0);
#else
  *(uint4*)lds_dst = *(const uint4*)gsrc;
#endif
}

__device__ __forceinline__ void wait_async_le1() {
#ifdef HAVE_ASYNC_LDS
  __builtin_amdgcn_s_wait_asynccnt(1);
#endif
}
__device__ __forceinline__ void wait_async_le0() {
#ifdef HAVE_ASYNC_LDS
  __builtin_amdgcn_s_wait_asynccnt(0);
#endif
}

// ---------------------------------------------------------------------------
// A fragment (16x32, MxK), contiguous k order shared with the B loaders:
//   lane L (L<16): row M = r0+L,   K=0..15 (contiguous)
//   lane L (>=16): row M = r0+L-16, K=16..31 (contiguous)
__device__ __forceinline__ v16h load_tile_a(const _Float16* __restrict__ base,
                                            int ld, int r0, int c0) {
  const int lane = threadIdx.x & 31;
  const _Float16* p = base + (size_t)(r0 + (lane & 15)) * ld
                           + c0 + ((lane >> 4) << 4);
  TileU t;
  t.q[0] = *(const uint4*)(p);
  t.q[1] = *(const uint4*)(p + 8);
  return t.v;
}

// B fragment (32x16 KxN) from global: lane holds column n = n0+(L&15);
// lo lanes K=k0..k0+15, hi lanes K=k0+16..k0+31 (contiguous in k).
__device__ __forceinline__ v16h load_tile_b(const _Float16* __restrict__ W,
                                            int ld, int n0, int k0) {
  const int lane = threadIdx.x & 31;
  const _Float16* p = W + (size_t)(n0 + (lane & 15)) * ld
                        + k0 + ((lane >> 4) << 4);
  TileU t;
  t.q[0] = *(const uint4*)(p);
  t.q[1] = *(const uint4*)(p + 8);
  return t.v;
}

// Same B fragment but from the staged LDS panel (rows = gate column n,
// BROW_STRIDE halves apart) -> ds_load_b128 x2, bank-conflict-free.
__device__ __forceinline__ v16h load_tile_b_lds(const _Float16* buf, int nt) {
  const int lane = threadIdx.x & 31;
  const _Float16* p = buf + (nt * 16 + (lane & 15)) * BROW_STRIDE
                          + ((lane >> 4) << 4);
  TileU t;
  t.q[0] = *(const uint4*)(p);
  t.q[1] = *(const uint4*)(p + 8);
  return t.v;
}

__device__ __forceinline__ v8f wmma_f16(v16h a, v16h b, v8f c) {
  // (neg_a, A, neg_b, B, c_mod, C, reuse_a, reuse_b)
  return __builtin_amdgcn_wmma_f32_16x16x32_f16(false, a, false, b,
                                                (short)0, c, false, false);
}

// C/D tile store: VGPR r -> M = r + 8*(lane>=16), N = lane&15
__device__ __forceinline__ void store_tile(float* __restrict__ base, int ld,
                                           int r0, int c0, v8f c) {
  const int lane = threadIdx.x & 31;
  const int m = r0 + ((lane >> 4) << 3);
  const int n = c0 + (lane & 15);
#pragma unroll
  for (int r = 0; r < 8; ++r) base[(size_t)(m + r) * ld + n] = c[r];
}

// ---------------------------------------------------------------------------
// Grid-wide sense-reversal barrier over all NBLK persistent workgroups.
__device__ __forceinline__ void grid_barrier(unsigned* bar) {
  __syncthreads();
  if (threadIdx.x == 0) {
    __threadfence();
    unsigned gen = __hip_atomic_load(&bar[1], __ATOMIC_ACQUIRE,
                                     __HIP_MEMORY_SCOPE_AGENT);
    unsigned arrived = __hip_atomic_fetch_add(&bar[0], 1u, __ATOMIC_ACQ_REL,
                                              __HIP_MEMORY_SCOPE_AGENT);
    if (arrived == NBLK - 1) {
      __hip_atomic_store(&bar[0], 0u, __ATOMIC_RELAXED,
                         __HIP_MEMORY_SCOPE_AGENT);
      __threadfence();
      __hip_atomic_store(&bar[1], gen + 1u, __ATOMIC_RELEASE,
                         __HIP_MEMORY_SCOPE_AGENT);
    } else {
      while (__hip_atomic_load(&bar[1], __ATOMIC_ACQUIRE,
                               __HIP_MEMORY_SCOPE_AGENT) == gen) {
        __builtin_amdgcn_s_sleep(1);
      }
    }
    __threadfence();
  }
  __syncthreads();
}

// ---------------------------------------------------------------------------
// Gate GEMM: gates[128 x 2048] = A[128 x K] * W^T (W row-major [2048 x K], f16)
// Workgroup owns 64 gate columns; B panels double-buffered through LDS via
// async DMA (one global_load_async_to_lds_b128 per thread per panel = 4KB).
// Each of the 8 waves owns one 16-row M tile and consumes the shared panel.
__device__ void gemm_gates(const _Float16* __restrict__ A, int K,
                           const _Float16* __restrict__ W,
                           float* __restrict__ gates,
                           _Float16* __restrict__ Bbuf /* [2*BPANEL] LDS */) {
  const int m0 = (threadIdx.x >> 5) * 16;      // wave id * 16 -> rows
  const int n0 = blockIdx.x * 64;              // 64 columns per workgroup
  // per-thread staging slot: 256 threads x 16B = one 64x32 f16 panel
  const int brow = threadIdx.x >> 2;           // 0..63 (gate column within slice)
  const int bcol = (threadIdx.x & 3) * 8;      // f16 offset within row
  const _Float16* gsrc = W + (size_t)(n0 + brow) * K + bcol;
  _Float16* ldst = Bbuf + brow * BROW_STRIDE + bcol;

  copy_b128_to_lds(ldst, gsrc);                // stage panel k0=0 into buf 0
  v8f acc0 = {}, acc1 = {}, acc2 = {}, acc3 = {};
  int cur = 0;
#pragma unroll 1
  for (int k0 = 0; k0 < K; k0 += 32) {
    if (k0 + 32 < K) {                         // stage next panel (overlapped)
      copy_b128_to_lds(ldst + (cur ^ 1) * BPANEL, gsrc + k0 + 32);
      wait_async_le1();                        // current panel landed
    } else {
      wait_async_le0();
    }
    __syncthreads();                           // panel visible to all waves
    __builtin_prefetch((const void*)(A + (size_t)(m0 + (threadIdx.x & 15)) * K
                                       + k0 + 32), 0, 3);
    v16h a  = load_tile_a(A, K, m0, k0);
    const _Float16* bp = Bbuf + cur * BPANEL;
    v16h b0 = load_tile_b_lds(bp, 0);
    v16h b1 = load_tile_b_lds(bp, 1);
    v16h b2 = load_tile_b_lds(bp, 2);
    v16h b3 = load_tile_b_lds(bp, 3);
    acc0 = wmma_f16(a, b0, acc0);
    acc1 = wmma_f16(a, b1, acc1);
    acc2 = wmma_f16(a, b2, acc2);
    acc3 = wmma_f16(a, b3, acc3);
    __syncthreads();                           // done reading before overwrite
    cur ^= 1;
  }
  store_tile(gates, 4 * HID, m0, n0 +  0, acc0);
  store_tile(gates, 4 * HID, m0, n0 + 16, acc1);
  store_tile(gates, 4 * HID, m0, n0 + 32, acc2);
  store_tile(gates, 4 * HID, m0, n0 + 48, acc3);
}

// LSTM cell elementwise: consume gates[128 x 2048] (order f,i,c,o), update C,
// produce h (f32 copy + f16 into the combined activation buffers).
__device__ void lstm_elemwise(const float* __restrict__ gates,
                              const float* __restrict__ bg,
                              float* __restrict__ C, float* __restrict__ hf32,
                              _Float16* __restrict__ hdst0, int ld0, int off0,
                              _Float16* __restrict__ hdst1, int ld1, int off1) {
  const int tid = blockIdx.x * blockDim.x + threadIdx.x;   // 0..8191
#pragma unroll
  for (int r = 0; r < 8; ++r) {
    const int e = r * NTHREADS_TOTAL + tid;   // 0..65535, coalesced in j
    const int b = e >> 9;                     // batch row
    const int j = e & (HID - 1);              // hidden index
    const float* g = gates + (size_t)b * (4 * HID);
    float f  = g[j]             + bg[j];
    float i_ = g[HID + j]       + bg[HID + j];
    float cc = g[2 * HID + j]   + bg[2 * HID + j];
    float o  = g[3 * HID + j]   + bg[3 * HID + j];
    float ft = 1.0f / (1.0f + __expf(-f));
    float it = 1.0f / (1.0f + __expf(-i_));
    float ct = tanhf(cc);
    float Cn = ft * C[e] + it * ct;
    C[e] = Cn;
    float ot = 1.0f / (1.0f + __expf(-o));
    float h  = ot * tanhf(Cn);
    hf32[e] = h;
    _Float16 hh = (_Float16)h;
    hdst0[(size_t)b * ld0 + off0 + j] = hh;
    if (hdst1) hdst1[(size_t)b * ld1 + off1 + j] = hh;
  }
}

// out_t[128 x 256] = h1[128 x 512] * W_out^T + b_out ; h1 lives in
// combined1[:, HID:2*HID]. 128 16x16 tiles -> first 128 waves take one each.
__device__ void out_proj(const _Float16* __restrict__ combined1,
                         const _Float16* __restrict__ Wouth,
                         const float* __restrict__ b_out,
                         float* __restrict__ out_t) {
  const int gwave = blockIdx.x * (NTHR / 32) + (threadIdx.x >> 5);
  if (gwave >= (BATCH / 16) * (OUT_DIM / 16)) return;
  const int m0 = (gwave >> 4) * 16;
  const int n0 = (gwave & 15) * 16;
  v8f acc = {};
#pragma unroll 1
  for (int k0 = 0; k0 < HID; k0 += 32) {
    v16h a = load_tile_a(combined1, 2 * HID, m0, HID + k0);
    v16h b = load_tile_b(Wouth, HID, n0, k0);
    acc = wmma_f16(a, b, acc);
  }
  const int lane = threadIdx.x & 31;
  const int m = m0 + ((lane >> 4) << 3);
  const int n = n0 + (lane & 15);
  const float bias = b_out[n];
#pragma unroll
  for (int r = 0; r < 8; ++r)
    out_t[(size_t)(m + r) * OUT_DIM + n] = acc[r] + bias;
}

// Convert x_t (f32) into combined0[:, 0:IN_DIM] (f16, ld=IN_DIM+HID)
__device__ void convert_x(const float* __restrict__ x, int t,
                          _Float16* __restrict__ combined0) {
  const int tid = blockIdx.x * blockDim.x + threadIdx.x;
  const float* xt = x + (size_t)t * BATCH * IN_DIM;
#pragma unroll
  for (int r = 0; r < 4; ++r) {
    const int e = r * NTHREADS_TOTAL + tid;   // 0..32767
    const int b = e >> 8;
    const int k = e & (IN_DIM - 1);
    combined0[(size_t)b * (IN_DIM + HID) + k] = (_Float16)xt[e];
  }
}

// ---------------------------------------------------------------------------
__global__ void __launch_bounds__(NTHR)
lstm_scan_kernel(const float* __restrict__ x,
                 const float* __restrict__ bg0, const float* __restrict__ bg1,
                 const float* __restrict__ b_out,
                 const _Float16* __restrict__ Wg0h,
                 const _Float16* __restrict__ Wg1h,
                 const _Float16* __restrict__ Wouth,
                 _Float16* __restrict__ combined0,   // [128 x 768]  f16
                 _Float16* __restrict__ combined1,   // [128 x 1024] f16
                 float* __restrict__ gates,          // [128 x 2048] f32
                 float* __restrict__ C0, float* __restrict__ C1,
                 float* __restrict__ h0f32, float* __restrict__ h1f32,
                 unsigned* __restrict__ bar,
                 float* __restrict__ out) {
  __shared__ __align__(16) _Float16 Bbuf[2 * BPANEL];  // 10 KB double buffer
  convert_x(x, 0, combined0);
#pragma unroll 1
  for (int t = 0; t < SEQ; ++t) {
    grid_barrier(bar);
    // layer 0 gates: combined0 = [x_t | h0], K = 768
    gemm_gates(combined0, IN_DIM + HID, Wg0h, gates, Bbuf);
    grid_barrier(bar);
    // cell 0: h0 -> combined0[:,256:768] and combined1[:,0:512]
    lstm_elemwise(gates, bg0, C0, h0f32,
                  combined0, IN_DIM + HID, IN_DIM,
                  combined1, 2 * HID, 0);
    grid_barrier(bar);
    // layer 1 gates: combined1 = [h0 | h1], K = 1024
    gemm_gates(combined1, 2 * HID, Wg1h, gates, Bbuf);
    grid_barrier(bar);
    // cell 1: h1 -> combined1[:,512:1024]
    lstm_elemwise(gates, bg1, C1, h1f32,
                  combined1, 2 * HID, HID,
                  (_Float16*)nullptr, 0, 0);
    if (t + 1 < SEQ) convert_x(x, t + 1, combined0);  // safe: 2 barriers away
    grid_barrier(bar);
    out_proj(combined1, Wouth, b_out, out + (size_t)t * BATCH * OUT_DIM);
  }
  grid_barrier(bar);
  // Final states appended after outputs: h_0, C_0, h_1, C_1 (each 128x512 f32)
  float* tail = out + (size_t)SEQ * BATCH * OUT_DIM;
  const int tid = blockIdx.x * blockDim.x + threadIdx.x;
  for (int e = tid; e < BATCH * HID; e += NTHREADS_TOTAL) {
    tail[e]                    = h0f32[e];
    tail[BATCH * HID + e]      = C0[e];
    tail[2 * BATCH * HID + e]  = h1f32[e];
    tail[3 * BATCH * HID + e]  = C1[e];
  }
}

__global__ void init_ws_kernel(unsigned* __restrict__ bar,
                               _Float16* __restrict__ combined0,
                               _Float16* __restrict__ combined1,
                               float* __restrict__ C0, float* __restrict__ C1,
                               float* __restrict__ h0f32,
                               float* __restrict__ h1f32) {
  const int tid = blockIdx.x * blockDim.x + threadIdx.x;
  const int stride = gridDim.x * blockDim.x;
  if (tid < 64) bar[tid] = 0u;
  for (int e = tid; e < BATCH * (IN_DIM + HID); e += stride)
    combined0[e] = (_Float16)0.0f;
  for (int e = tid; e < BATCH * 2 * HID; e += stride)
    combined1[e] = (_Float16)0.0f;
  for (int e = tid; e < BATCH * HID; e += stride) {
    C0[e] = 0.0f; C1[e] = 0.0f; h0f32[e] = 0.0f; h1f32[e] = 0.0f;
  }
}

__global__ void convert_w_kernel(const float* __restrict__ Wg0,
                                 const float* __restrict__ Wg1,
                                 const float* __restrict__ Wout,
                                 _Float16* __restrict__ Wg0h,
                                 _Float16* __restrict__ Wg1h,
                                 _Float16* __restrict__ Wouth) {
  const int tid = blockIdx.x * blockDim.x + threadIdx.x;
  const int stride = gridDim.x * blockDim.x;
  for (int e = tid; e < 4 * HID * (IN_DIM + HID); e += stride)
    Wg0h[e] = (_Float16)Wg0[e];
  for (int e = tid; e < 4 * HID * 2 * HID; e += stride)
    Wg1h[e] = (_Float16)Wg1[e];
  for (int e = tid; e < OUT_DIM * HID; e += stride)
    Wouth[e] = (_Float16)Wout[e];
}

// ---------------------------------------------------------------------------
extern "C" void kernel_launch(void* const* d_in, const int* in_sizes, int n_in,
                              void* d_out, int out_size, void* d_ws,
                              size_t ws_size, hipStream_t stream) {
  (void)in_sizes; (void)n_in; (void)out_size; (void)ws_size;
  const float* x     = (const float*)d_in[0];
  const float* Wg0   = (const float*)d_in[1];
  const float* bg0   = (const float*)d_in[2];
  const float* Wg1   = (const float*)d_in[3];
  const float* bg1   = (const float*)d_in[4];
  const float* Wout  = (const float*)d_in[5];
  const float* b_out = (const float*)d_in[6];
  float* out = (float*)d_out;

  // Workspace carve (256B aligned regions, ~10.2 MB total)
  char* ws = (char*)d_ws;
  size_t off = 0;
  auto carve = [&](size_t bytes) -> void* {
    off = (off + 255) & ~(size_t)255;
    void* p = ws + off;
    off += bytes;
    return p;
  };
  unsigned* bar       = (unsigned*)carve(256);
  _Float16* Wg0h      = (_Float16*)carve((size_t)4 * HID * (IN_DIM + HID) * 2);
  _Float16* Wg1h      = (_Float16*)carve((size_t)4 * HID * 2 * HID * 2);
  _Float16* Wouth     = (_Float16*)carve((size_t)OUT_DIM * HID * 2);
  _Float16* combined0 = (_Float16*)carve((size_t)BATCH * (IN_DIM + HID) * 2);
  _Float16* combined1 = (_Float16*)carve((size_t)BATCH * 2 * HID * 2);
  float*    gates     = (float*)carve((size_t)BATCH * 4 * HID * 4);
  float*    C0        = (float*)carve((size_t)BATCH * HID * 4);
  float*    C1        = (float*)carve((size_t)BATCH * HID * 4);
  float*    h0f32     = (float*)carve((size_t)BATCH * HID * 4);
  float*    h1f32     = (float*)carve((size_t)BATCH * HID * 4);

  init_ws_kernel<<<64, 256, 0, stream>>>(bar, combined0, combined1,
                                         C0, C1, h0f32, h1f32);
  convert_w_kernel<<<512, 256, 0, stream>>>(Wg0, Wg1, Wout,
                                            Wg0h, Wg1h, Wouth);
  lstm_scan_kernel<<<NBLK, NTHR, 0, stream>>>(x, bg0, bg1, b_out,
                                              Wg0h, Wg1h, Wouth,
                                              combined0, combined1, gates,
                                              C0, C1, h0f32, h1f32,
                                              bar, out);
}
